// RNN_80195629351378
// MI455X (gfx1250) — matrix operand order repaired and checked by reading, not compiled, not run on hardware
//
#include <hip/hip_runtime.h>
#include <hip/hip_bf16.h>

#if __has_include(<hip/amd_detail/amd_gfx1250_TDM.h>)
#define TDM_SIX_ARGS 1
#else
#define TDM_SIX_ARGS 0
#endif

// CDNA5 / gfx1250 WMMA vector types
typedef __attribute__((ext_vector_type(16))) __bf16 v16bf;
typedef __attribute__((ext_vector_type(8)))  __bf16 v8bf;
typedef __attribute__((ext_vector_type(8)))  float  v8f;
typedef __attribute__((ext_vector_type(4)))  unsigned int u32x4;
typedef __attribute__((ext_vector_type(8)))  int i32x8;
typedef __attribute__((ext_vector_type(4)))  int i32x4;

constexpr int N_STEPS = 16384;
constexpr int DIN     = 128;
constexpr int NH      = 1024;
constexpr int DOUT    = 128;
constexpr float NEG_SLOPE = 0.01f;

// Cooperative-scan geometry: 16 blocks, each owns a 64-column slice whose
// recurrent weights (2 x 128 KB bf16) live in its WGP's 320 KB LDS.
constexpr int COLS = 64;
constexpr int NBLK = NH / COLS;                    // 16
constexpr int SCAN_THREADS = 512;                  // 16 wave32 waves
constexpr size_t SZ_W = (size_t)COLS * NH * 2;     // 131072 B per weight slice
constexpr size_t LDS_TOTAL = 2 * SZ_W + 2048 + 2048 + 1024 + 256;

// ---------------------------------------------------------------------------
// WMMA operand loaders.
// 16-bit A 16x32: lane L -> row M=L%16; halves e -> K=e+16g
// (g=L/16)  => two contiguous 16B loads per lane.
__device__ __forceinline__ v16bf load_a_tile(const __bf16* base, int ld,
                                             int lane) {
  const int m = lane & 15, g = lane >> 4;
  const __bf16* p = base + (size_t)m * ld + 16 * g;
  v8bf lo = *(const v8bf*)(p);
  v8bf hi = *(const v8bf*)(p + 8);
  v16bf a;
#pragma unroll
  for (int i = 0; i < 8; ++i) { a[i] = lo[i]; a[i + 8] = hi[i]; }
  return a;
}

// 16-bit B 32x16: lane L -> col N=L%16; halves e -> K=e+16g  => one 32B load.
__device__ __forceinline__ v16bf load_b_tile(const __bf16* baseT, int ld,
                                             int lane) {
  const int n = lane & 15, g = lane >> 4;
  return *(const v16bf*)(baseT + (size_t)n * ld + 16 * g);
}

__device__ __forceinline__ v8f wmma_bf16(v16bf a, v16bf b, v8f c) {
  return __builtin_amdgcn_wmma_f32_16x16x32_bf16(false, a, false, b, (short)0,
                                                 c, false, false);
}

// ---------------------------------------------------------------------------
// Tensor Data Mover: 1-D contiguous tile, data_size=3 (8B units), count=1,
// type=2 ("image").  D# layout per CDNA5 ISA 8.3/8.4.
__device__ __forceinline__ void tdm_load_1d(unsigned lds_byte_off,
                                            unsigned long long gaddr,
                                            unsigned units8B) {
#if __has_builtin(__builtin_amdgcn_tensor_load_to_lds)
  u32x4 g0;
  g0[0] = 1u;                                   // count=1, user descriptor
  g0[1] = lds_byte_off;                         // lds_addr
  g0[2] = (unsigned)gaddr;                      // global_addr[31:0]
  g0[3] = (unsigned)((gaddr >> 32) & 0x01FFFFFFu) | (2u << 30);  // addr|type
  i32x8 g1;
  g1[0] = 0x30000;                              // data_size=3 (8B), no mask
  g1[1] = (int)(units8B << 16);                 // tensor_dim0[15:0]
  g1[2] = (int)(1u << 16);                      // tensor_dim0 hi=0, dim1=1
  g1[3] = (int)(units8B << 16);                 // tile_dim0
  g1[4] = 0;                                    // tile_dim1=0 (1-D), dim2=0
  g1[5] = (int)units8B;                         // tensor_dim0_stride
  g1[6] = 0;
  g1[7] = 0;
  i32x4 z4 = {0, 0, 0, 0};
#if TDM_SIX_ARGS
  i32x8 z8 = {0, 0, 0, 0, 0, 0, 0, 0};
  __builtin_amdgcn_tensor_load_to_lds(g0, g1, z4, z4, z8, 0);
#else
  __builtin_amdgcn_tensor_load_to_lds(g0, g1, z4, z4, 0);
#endif
#else
  (void)lds_byte_off; (void)gaddr; (void)units8B;
#endif
}

// ---------------------------------------------------------------------------
// Agent-scope sense-reversing barrier across NBLK co-resident blocks.
__device__ __forceinline__ void grid_barrier(unsigned* counter, unsigned* gen) {
  __syncthreads();
  if (threadIdx.x == 0) {
    unsigned g = __hip_atomic_load(gen, __ATOMIC_ACQUIRE,
                                   __HIP_MEMORY_SCOPE_AGENT);
    unsigned arrived = __hip_atomic_fetch_add(counter, 1u, __ATOMIC_ACQ_REL,
                                              __HIP_MEMORY_SCOPE_AGENT);
    if (arrived == (unsigned)(NBLK - 1)) {
      __hip_atomic_store(counter, 0u, __ATOMIC_RELAXED,
                         __HIP_MEMORY_SCOPE_AGENT);
      __hip_atomic_store(gen, g + 1u, __ATOMIC_RELEASE,
                         __HIP_MEMORY_SCOPE_AGENT);
    } else {
      while (__hip_atomic_load(gen, __ATOMIC_ACQUIRE,
                               __HIP_MEMORY_SCOPE_AGENT) == g)
        __builtin_amdgcn_s_sleep(1);
    }
  }
  __syncthreads();
}

// ---------------------------------------------------------------------------
// Prep kernels.
__global__ void k_cvt_bf16(const float* __restrict__ src,
                           __bf16* __restrict__ dst, int n) {
  int i = blockIdx.x * blockDim.x + threadIdx.x;
  if (i < n) dst[i] = (__bf16)src[i];
}

__global__ void k_transpose_bf16(const float* __restrict__ src,
                                 __bf16* __restrict__ dst, int R, int C) {
  int i = blockIdx.x * blockDim.x + threadIdx.x;
  if (i >= R * C) return;
  int r = i / C, c = i % C;
  dst[(size_t)c * R + r] = (__bf16)src[(size_t)r * C + c];
}

__global__ void k_init(const float* __restrict__ h0, __bf16* __restrict__ hbuf,
                       unsigned* __restrict__ bar) {
  int i = blockIdx.x * blockDim.x + threadIdx.x;
  if (i < NH) hbuf[i] = (__bf16)h0[i];
  if (i < 2) bar[i] = 0u;
}

// ---------------------------------------------------------------------------
// Phase 1: Xpre = X @ W1x + b1   (one wave per 16x16 tile, K=128)
__global__ __launch_bounds__(256) void k_xpre(const __bf16* __restrict__ Xbf,
                                              const __bf16* __restrict__ W1xT,
                                              const float* __restrict__ b1,
                                              float* __restrict__ Xpre) {
  const int lane = threadIdx.x & 31;
  const int wg   = blockIdx.x * (blockDim.x >> 5) + (threadIdx.x >> 5);
  const int m0   = (wg >> 6) << 4;
  const int j0   = (wg & 63) << 4;
  v8f acc = {};
#pragma unroll
  for (int k0 = 0; k0 < DIN; k0 += 32) {
    v16bf a = load_a_tile(Xbf + (size_t)m0 * DIN + k0, DIN, lane);
    v16bf b = load_b_tile(W1xT + (size_t)j0 * DIN + k0, DIN, lane);
    acc = wmma_bf16(a, b, acc);
  }
  const int g = lane >> 4, ln = lane & 15;
  const float bias = b1[j0 + ln];
#pragma unroll
  for (int r = 0; r < 8; ++r)
    Xpre[(size_t)(m0 + r + 8 * g) * NH + j0 + ln] = acc[r] + bias;
}

// ---------------------------------------------------------------------------
// Phase 2: cooperative scan.  16 blocks x 512 threads; block b owns columns
// [64b, 64b+64).  Recurrent weight slices TDM-staged into LDS once; per step
// only the 1024-wide activation vector crosses WGPs (through L2).
__global__ __launch_bounds__(SCAN_THREADS) void k_scan(
    const float* __restrict__ Xpre, const __bf16* __restrict__ W1hT,
    const __bf16* __restrict__ W2T, const float* __restrict__ b2,
    __bf16* __restrict__ Hn, float* __restrict__ h_final,
    __bf16* __restrict__ zbuf, __bf16* __restrict__ hbuf,
    unsigned* __restrict__ bar) {
  extern __shared__ __align__(32) char smem[];
  __bf16* sW1 = (__bf16*)(smem);                    // 128 KB: W1hT slice
  __bf16* sW2 = (__bf16*)(smem + SZ_W);             // 128 KB: W2T slice
  __bf16* sh  = (__bf16*)(smem + 2 * SZ_W);         // full hidden (bf16)
  __bf16* sz  = (__bf16*)(smem + 2 * SZ_W + 2048);  // full z (bf16)
  float*  part = (float*)(smem + 2 * SZ_W + 4096);  // 16x16 partial sums
  float*  sb2  = (float*)(smem + 2 * SZ_W + 5120);  // bias slice

  const int tid  = threadIdx.x;
  const int lane = tid & 31;
  const int wave = tid >> 5;
  const int g = lane >> 4, ln = lane & 15;
  const int strip = wave & 3;       // 4 strips of 16 output columns
  const int chunk = wave >> 2;      // 4-way K split (256 each)
  const int colbase = blockIdx.x * COLS;

  // --- Stage weights into LDS (TDM async DMA; plain-load fallback) ---
#if __has_builtin(__builtin_amdgcn_tensor_load_to_lds)
  if (wave == 0) {
    tdm_load_1d((unsigned)(size_t)sW1,
                (unsigned long long)(size_t)(W1hT + (size_t)colbase * NH),
                (unsigned)(SZ_W / 8));
    tdm_load_1d((unsigned)(size_t)sW2,
                (unsigned long long)(size_t)(W2T + (size_t)colbase * NH),
                (unsigned)(SZ_W / 8));
    __builtin_amdgcn_s_wait_tensorcnt(0);
  }
#else
  for (size_t i = tid; i < SZ_W / 8; i += SCAN_THREADS) {
    ((unsigned long long*)sW1)[i] =
        ((const unsigned long long*)(W1hT + (size_t)colbase * NH))[i];
    ((unsigned long long*)sW2)[i] =
        ((const unsigned long long*)(W2T + (size_t)colbase * NH))[i];
  }
#endif
  sh[tid] = hbuf[tid];
  sh[tid + 512] = hbuf[tid + 512];
  if (tid < COLS) sb2[tid] = b2[colbase + tid];
  __syncthreads();

#pragma unroll 1
  for (int t = 0; t < N_STEPS; ++t) {
    if (tid < COLS && t + 1 < N_STEPS)  // global_prefetch_b8 next Xpre row
      __builtin_prefetch(Xpre + (size_t)(t + 1) * NH + colbase + tid, 0, 1);

    // --- GEMV1 partials: y = h @ W1h (our 64 cols; K split across waves) ---
    {
      const int j0 = strip << 4, k0b = chunk << 8;
      v8f acc = {};
#pragma unroll 2
      for (int k0 = k0b; k0 < k0b + 256; k0 += 32) {
        v16bf a = load_a_tile(sW1 + (size_t)j0 * NH + k0, NH, lane);
        v16bf b = *(const v16bf*)(sh + k0 + 16 * g);
        acc = wmma_bf16(a, b, acc);
      }
      if (ln == 0) {
#pragma unroll
        for (int r = 0; r < 8; ++r)
          part[(((strip << 2) | chunk) << 4) + 8 * g + r] = acc[r];
      }
    }
    __syncthreads();
    if (tid < COLS) {  // reduce K partials, tanh, publish z slice
      const int st = tid >> 4, jn = tid & 15;
      float s = Xpre[(size_t)t * NH + colbase + tid];
#pragma unroll
      for (int c = 0; c < 4; ++c) s += part[(((st << 2) | c) << 4) + jn];
      zbuf[colbase + tid] = (__bf16)tanhf(s);
      __threadfence();
    }
    grid_barrier(bar, bar + 1);
    sz[tid] = zbuf[tid];
    sz[tid + 512] = zbuf[tid + 512];
    __syncthreads();

    // --- GEMV2 partials: y = z @ W2 ---
    {
      const int j0 = strip << 4, k0b = chunk << 8;
      v8f acc = {};
#pragma unroll 2
      for (int k0 = k0b; k0 < k0b + 256; k0 += 32) {
        v16bf a = load_a_tile(sW2 + (size_t)j0 * NH + k0, NH, lane);
        v16bf b = *(const v16bf*)(sz + k0 + 16 * g);
        acc = wmma_bf16(a, b, acc);
      }
      if (ln == 0) {
#pragma unroll
        for (int r = 0; r < 8; ++r)
          part[(((strip << 2) | chunk) << 4) + 8 * g + r] = acc[r];
      }
    }
    __syncthreads();
    if (tid < COLS) {  // reduce, bias + leaky_relu, publish new hidden
      const int st = tid >> 4, jn = tid & 15;
      float s = sb2[tid];
#pragma unroll
      for (int c = 0; c < 4; ++c) s += part[(((st << 2) | c) << 4) + jn];
      s = (s >= 0.f) ? s : NEG_SLOPE * s;
      __bf16 hb = (__bf16)s;
      hbuf[colbase + tid] = hb;
      Hn[(size_t)t * NH + colbase + tid] = hb;
      if (t == N_STEPS - 1) h_final[colbase + tid] = s;
      __threadfence();
    }
    grid_barrier(bar, bar + 1);
    sh[tid] = hbuf[tid];
    sh[tid + 512] = hbuf[tid + 512];
    __syncthreads();
  }
}

// ---------------------------------------------------------------------------
// Phase 3: Out = Hn @ Wo + bo   (one wave per 16x16 tile, K=1024)
__global__ __launch_bounds__(256) void k_readout(const __bf16* __restrict__ Hn,
                                                 const __bf16* __restrict__ WoT,
                                                 const float* __restrict__ bo,
                                                 float* __restrict__ Out) {
  const int lane = threadIdx.x & 31;
  const int wg   = blockIdx.x * (blockDim.x >> 5) + (threadIdx.x >> 5);
  const int m0   = (wg >> 3) << 4;
  const int n0   = (wg & 7) << 4;
  v8f acc = {};
#pragma unroll 4
  for (int k0 = 0; k0 < NH; k0 += 32) {
    v16bf a = load_a_tile(Hn + (size_t)m0 * NH + k0, NH, lane);
    v16bf b = load_b_tile(WoT + (size_t)n0 * NH + k0, NH, lane);
    acc = wmma_bf16(a, b, acc);
  }
  const int g = lane >> 4, ln = lane & 15;
  const float bias = bo[n0 + ln];
#pragma unroll
  for (int r = 0; r < 8; ++r)
    Out[(size_t)(m0 + r + 8 * g) * DOUT + n0 + ln] = acc[r] + bias;
}

// ---------------------------------------------------------------------------
extern "C" void kernel_launch(void* const* d_in, const int* in_sizes, int n_in,
                              void* d_out, int out_size, void* d_ws,
                              size_t ws_size, hipStream_t stream) {
  (void)in_sizes; (void)n_in; (void)out_size; (void)ws_size;
  const float* x  = (const float*)d_in[0];
  const float* h0 = (const float*)d_in[1];
  const float* W1 = (const float*)d_in[2];
  const float* b1 = (const float*)d_in[3];
  const float* W2 = (const float*)d_in[4];
  const float* b2 = (const float*)d_in[5];
  const float* Wo = (const float*)d_in[6];
  const float* bo = (const float*)d_in[7];
  float* out = (float*)d_out;  // [N_STEPS*DOUT outputs] ++ [NH h_final]

  char*   ws   = (char*)d_ws;
  size_t  off  = 0;
  __bf16* Xbf  = (__bf16*)(ws + off); off += (size_t)N_STEPS * DIN * 2;
  __bf16* W1xT = (__bf16*)(ws + off); off += (size_t)NH * DIN * 2;
  __bf16* W1hT = (__bf16*)(ws + off); off += (size_t)NH * NH * 2;
  __bf16* W2T  = (__bf16*)(ws + off); off += (size_t)NH * NH * 2;
  __bf16* WoT  = (__bf16*)(ws + off); off += (size_t)DOUT * NH * 2;
  float*  Xpre = (float*)(ws + off);  off += (size_t)N_STEPS * NH * 4;
  __bf16* Hn   = (__bf16*)(ws + off); off += (size_t)N_STEPS * NH * 2;
  __bf16* zbuf = (__bf16*)(ws + off); off += (size_t)NH * 2;
  __bf16* hbuf = (__bf16*)(ws + off); off += (size_t)NH * 2;
  unsigned* bar = (unsigned*)(ws + off); off += 64;

  // Prep: bf16 convert + transposed weight layouts + barrier/hidden init
  k_cvt_bf16<<<(N_STEPS * DIN + 255) / 256, 256, 0, stream>>>(x, Xbf,
                                                              N_STEPS * DIN);
  k_transpose_bf16<<<(DIN * NH + 255) / 256, 256, 0, stream>>>(W1, W1xT, DIN, NH);
  k_transpose_bf16<<<(NH * NH + 255) / 256, 256, 0, stream>>>(
      W1 + (size_t)DIN * NH, W1hT, NH, NH);
  k_transpose_bf16<<<(NH * NH + 255) / 256, 256, 0, stream>>>(W2, W2T, NH, NH);
  k_transpose_bf16<<<(NH * DOUT + 255) / 256, 256, 0, stream>>>(Wo, WoT, NH, DOUT);
  k_init<<<(NH + 255) / 256, 256, 0, stream>>>(h0, hbuf, bar);

  // Phase 1: hoisted input GEMM
  k_xpre<<<(N_STEPS / 16) * (NH / 16) / 8, 256, 0, stream>>>(Xbf, W1xT, b1, Xpre);

  // Phase 2: cooperative LDS-resident recurrence (16 blocks, TDM staging)
  k_scan<<<NBLK, SCAN_THREADS, LDS_TOTAL, stream>>>(Xpre, W1hT, W2T, b2, Hn,
                                                    out + (size_t)N_STEPS * DOUT,
                                                    zbuf, hbuf, bar);

  // Phase 3: deferred readout GEMM
  k_readout<<<(N_STEPS / 16) * (DOUT / 16) / 8, 256, 0, stream>>>(Hn, WoT, bo, out);
}
